// MolecularAttentionPooling_39582418600426
// MI455X (gfx1250) — matrix operand order repaired and checked by reading, not compiled, not run on hardware
//
#include <hip/hip_runtime.h>
#include <hip/hip_bf16.h>

typedef __attribute__((ext_vector_type(16))) _Float16 v16h;
typedef __attribute__((ext_vector_type(8)))  _Float16 v8h;
typedef __attribute__((ext_vector_type(8)))  float    v8f;

#define GNUM 4096
#define EDIM 128
#define BN   64
#define APITCH 136   // f16 row pitch (272B = 17*16B -> conflict-free row-indexed b128)
#define CPITCH 132   // f32 row pitch

// ---- order-preserving float <-> uint encoding for atomic max on logits ----
__device__ __forceinline__ unsigned encf(float f) {
    unsigned u = __float_as_uint(f);
    return (u & 0x80000000u) ? ~u : (u | 0x80000000u);
}
__device__ __forceinline__ float decf(unsigned e) {
    unsigned u = (e & 0x80000000u) ? (e & 0x7FFFFFFFu) : ~e;
    return __uint_as_float(u);
}

// ---------------------------------------------------------------------------
// 64x128 = (64x128 f16 in LDS) x (128x128 f16 weights, row-major, as B=W^T)
// 8 waves: wave w -> row-tile w>>1 (16 rows), 4 col-tiles starting (w&1)*4.
// A fragment per ISA 16-bit A layout: lane(lo,hi): rows lo, K chunks
// {kb+hi*8 .. +7} and {kb+16+hi*8 .. +7}.
// B fragment: lane(lo,hi) = weight row (ct*16+lo), K chunks
// {kb+hi*8 .. +7} and {kb+16+hi*8 .. +7} (B[k][j] = W[j][k]), matching A.
// C layout: VGPR r, lane(lo,hi) -> row rt*16 + hi*8 + r, col ct*16 + lo.
// ---------------------------------------------------------------------------
__device__ __forceinline__ void wave_gemm_64x128(const _Float16 (&sA)[BN][APITCH],
                                                 const _Float16* __restrict__ Wg,
                                                 float (&sC)[BN][CPITCH]) {
    const int tid  = threadIdx.x;
    const int wv   = tid >> 5;
    const int lane = tid & 31;
    const int rt   = wv >> 1;
    const int ct0  = (wv & 1) * 4;
    const int lo   = lane & 15;
    const int hi   = lane >> 4;

    v8f c[4];
#pragma unroll
    for (int t = 0; t < 4; ++t) c[t] = (v8f){0,0,0,0,0,0,0,0};

#pragma unroll
    for (int kk = 0; kk < 4; ++kk) {
        const int kb = kk * 32;
        v8h a0 = *(const v8h*)&sA[rt * 16 + lo][kb + hi * 8];
        v8h a1 = *(const v8h*)&sA[rt * 16 + lo][kb + 16 + hi * 8];
        v16h a;
#pragma unroll
        for (int i = 0; i < 8; ++i) { a[i] = a0[i]; a[i + 8] = a1[i]; }
#pragma unroll
        for (int t = 0; t < 4; ++t) {
            const int row = (ct0 + t) * 16 + lo;
            v8h b0 = *(const v8h*)(Wg + row * EDIM + kb + hi * 8);
            v8h b1 = *(const v8h*)(Wg + row * EDIM + kb + 16 + hi * 8);
            v16h b;
#pragma unroll
            for (int i = 0; i < 8; ++i) { b[i] = b0[i]; b[i + 8] = b1[i]; }
            c[t] = __builtin_amdgcn_wmma_f32_16x16x32_f16(
                false, a, false, b, (short)0, c[t], false, false);
        }
    }
#pragma unroll
    for (int t = 0; t < 4; ++t) {
        const int col = (ct0 + t) * 16 + lo;
#pragma unroll
        for (int r = 0; r < 8; ++r) sC[rt * 16 + hi * 8 + r][col] = c[t][r];
    }
}

// ---------------------------------------------------------------------------
__global__ void compute_q_kernel(const float* __restrict__ query,
                                 const float* __restrict__ in_proj_w,
                                 const float* __restrict__ in_proj_b,
                                 float* __restrict__ qout) {
    int j = threadIdx.x;  // 128 threads
    float s = 0.f;
#pragma unroll 4
    for (int e = 0; e < EDIM; ++e) s += query[e] * in_proj_w[j * EDIM + e];
    qout[j] = (s + in_proj_b[j]) * 0.17677669529663687f;  // 1/sqrt(32)
}

__global__ void cvt_weights_kernel(const float* __restrict__ in_proj_w,
                                   const float* __restrict__ out_w,
                                   const float* __restrict__ proj_w,
                                   _Float16* __restrict__ wk, _Float16* __restrict__ wv,
                                   _Float16* __restrict__ w1, _Float16* __restrict__ w2) {
    int i = blockIdx.x * blockDim.x + threadIdx.x;
    if (i < EDIM * EDIM) {
        wk[i] = (_Float16)in_proj_w[EDIM * EDIM + i];
        wv[i] = (_Float16)in_proj_w[2 * EDIM * EDIM + i];
        w1[i] = (_Float16)out_w[i];
        w2[i] = (_Float16)proj_w[i];
    }
}

__global__ void init_ws_kernel(unsigned* __restrict__ m_enc,
                               float* __restrict__ denom,
                               float* __restrict__ num) {
    int i = blockIdx.x * blockDim.x + threadIdx.x;
    if (i < GNUM * 4) { m_enc[i] = 0x007FFFFFu; /* enc(-inf) */ denom[i] = 0.f; }
    if (i < GNUM * EDIM) num[i] = 0.f;
}

// ---------------------------------------------------------------------------
__global__ __launch_bounds__(256) void pass1_kernel(
    const float* __restrict__ x, const int* __restrict__ seg,
    const float* __restrict__ qv, const float* __restrict__ in_proj_b,
    const _Float16* __restrict__ wk,
    unsigned* __restrict__ m_enc, float* __restrict__ logits, int N) {
    __shared__ __attribute__((aligned(32))) _Float16 sX[BN][APITCH];
    __shared__ __attribute__((aligned(16))) float    sK[BN][CPITCH];
    __shared__ float sQ[EDIM];
    __shared__ float sBk[EDIM];

    const int tid = threadIdx.x;
    const int nb  = blockIdx.x * BN;

    if (tid < 128) sQ[tid] = qv[tid];
    else if (tid < 256) sBk[tid - 128] = in_proj_b[EDIM + (tid - 128)];

    for (int idx = tid; idx < BN * EDIM; idx += 256) {
        int n = idx >> 7, d = idx & 127;
        int g = nb + n;
        float val = (g < N) ? x[(long)g * EDIM + d] : 0.f;
        sX[n][d] = (_Float16)val;
    }
    __syncthreads();

    wave_gemm_64x128(sX, wk, sK);
    __syncthreads();

    // logits: 64 nodes x 4 heads = 256 dot products of length 32
    int n = tid >> 2, h = tid & 3;
    int g = nb + n;
    if (g < N) {
        float s = 0.f;
#pragma unroll
        for (int d = 0; d < 32; ++d)
            s += sQ[h * 32 + d] * (sK[n][h * 32 + d] + sBk[h * 32 + d]);
        logits[(long)g * 4 + h] = s;
        atomicMax(&m_enc[seg[g] * 4 + h], encf(s));
    }
}

// ---------------------------------------------------------------------------
__global__ __launch_bounds__(256) void pass2_kernel(
    const float* __restrict__ x, const int* __restrict__ seg,
    const float* __restrict__ in_proj_b, const _Float16* __restrict__ wv,
    const unsigned* __restrict__ m_enc, const float* __restrict__ logits,
    float* __restrict__ denom, float* __restrict__ num, int N) {
    __shared__ __attribute__((aligned(32))) _Float16 sX[BN][APITCH];
    __shared__ __attribute__((aligned(16))) float    sV[BN][CPITCH];
    __shared__ float sP[BN][4];
    __shared__ float sBv[EDIM];
    __shared__ int   sSeg[BN];

    const int tid = threadIdx.x;
    const int nb  = blockIdx.x * BN;

    if (tid < BN) { int g = nb + tid; sSeg[tid] = (g < N) ? seg[g] : -1; }
    if (tid >= 128 && tid < 256) sBv[tid - 128] = in_proj_b[2 * EDIM + (tid - 128)];

    for (int idx = tid; idx < BN * EDIM; idx += 256) {
        int n = idx >> 7, d = idx & 127;
        int g = nb + n;
        float val = (g < N) ? x[(long)g * EDIM + d] : 0.f;
        sX[n][d] = (_Float16)val;
    }
    __syncthreads();

    wave_gemm_64x128(sX, wv, sV);
    __syncthreads();

    {   // p = exp(logit - segment_max)
        int n = tid >> 2, h = tid & 3;
        int g = nb + n;
        float p = 0.f;
        if (g < N) {
            float l = logits[(long)g * 4 + h];
            float m = decf(m_enc[sSeg[n] * 4 + h]);
            p = __expf(l - m);
        }
        sP[n][h] = p;
    }
    __syncthreads();

    // segmented reduction over sorted ids: one atomicAdd per segment run
    if (tid < 128) {
        const int d = tid, h = d >> 5;
        float acc = 0.f; int cur = -1;
        for (int n = 0; n < BN; ++n) {
            int s = sSeg[n];
            if (s < 0) break;
            if (s != cur) {
                if (cur >= 0) atomicAdd(&num[(long)cur * EDIM + d], acc);
                acc = 0.f; cur = s;
            }
            acc += sP[n][h] * (sV[n][d] + sBv[d]);
        }
        if (cur >= 0) atomicAdd(&num[(long)cur * EDIM + d], acc);
    } else if (tid < 132) {
        const int h = tid - 128;
        float acc = 0.f; int cur = -1;
        for (int n = 0; n < BN; ++n) {
            int s = sSeg[n];
            if (s < 0) break;
            if (s != cur) {
                if (cur >= 0) atomicAdd(&denom[cur * 4 + h], acc);
                acc = 0.f; cur = s;
            }
            acc += sP[n][h];
        }
        if (cur >= 0) atomicAdd(&denom[cur * 4 + h], acc);
    }
}

// ---------------------------------------------------------------------------
__global__ __launch_bounds__(256) void pass3_kernel(
    const float* __restrict__ num, const float* __restrict__ denom,
    const _Float16* __restrict__ w1, const _Float16* __restrict__ w2,
    const float* __restrict__ out_b, const float* __restrict__ proj_b,
    float* __restrict__ out) {
    __shared__ __attribute__((aligned(32))) _Float16 sA[BN][APITCH];
    __shared__ __attribute__((aligned(16))) float    sC[BN][CPITCH];

    const int tid = threadIdx.x;
    const int gb  = blockIdx.x * BN;

    // pooled = num / max(denom, 1e-9) -> f16
    for (int idx = tid; idx < BN * EDIM; idx += 256) {
        int r = idx >> 7, d = idx & 127;
        int g = gb + r, h = d >> 5;
        float den = fmaxf(denom[g * 4 + h], 1e-9f);
        sA[r][d] = (_Float16)(num[(long)g * EDIM + d] / den);
    }
    __syncthreads();

    wave_gemm_64x128(sA, w1, sC);           // attn = pooled @ out_w^T
    __syncthreads();

    for (int idx = tid; idx < BN * EDIM; idx += 256) {
        int r = idx >> 7, d = idx & 127;
        sA[r][d] = (_Float16)(sC[r][d] + out_b[d]);
    }
    __syncthreads();

    wave_gemm_64x128(sA, w2, sC);           // out = attn @ proj_w^T
    __syncthreads();

    for (int idx = tid; idx < BN * EDIM; idx += 256) {
        int r = idx >> 7, d = idx & 127;
        out[(long)(gb + r) * EDIM + d] = sC[r][d] + proj_b[d];
    }
}

// ---------------------------------------------------------------------------
extern "C" void kernel_launch(void* const* d_in, const int* in_sizes, int n_in,
                              void* d_out, int out_size, void* d_ws, size_t ws_size,
                              hipStream_t stream) {
    const float* x         = (const float*)d_in[0];
    const float* query     = (const float*)d_in[1];
    const float* in_proj_w = (const float*)d_in[2];
    const float* in_proj_b = (const float*)d_in[3];
    const float* out_w     = (const float*)d_in[4];
    const float* out_b     = (const float*)d_in[5];
    const float* proj_w    = (const float*)d_in[6];
    const float* proj_b    = (const float*)d_in[7];
    const int*   seg       = (const int*)d_in[8];
    const int N = in_sizes[0] / EDIM;

    char* ws = (char*)d_ws;
    size_t off = 0;
    auto take = [&](size_t bytes) -> char* {
        char* p = ws + off;
        off = (off + bytes + 255) & ~(size_t)255;
        return p;
    };
    float*     qf     = (float*)take(EDIM * sizeof(float));
    unsigned*  m_enc  = (unsigned*)take((size_t)GNUM * 4 * sizeof(unsigned));
    float*     denom  = (float*)take((size_t)GNUM * 4 * sizeof(float));
    float*     numbuf = (float*)take((size_t)GNUM * EDIM * sizeof(float));
    float*     logits = (float*)take((size_t)N * 4 * sizeof(float));
    _Float16*  wk     = (_Float16*)take((size_t)EDIM * EDIM * sizeof(_Float16));
    _Float16*  wv     = (_Float16*)take((size_t)EDIM * EDIM * sizeof(_Float16));
    _Float16*  w1     = (_Float16*)take((size_t)EDIM * EDIM * sizeof(_Float16));
    _Float16*  w2     = (_Float16*)take((size_t)EDIM * EDIM * sizeof(_Float16));
    (void)ws_size; (void)n_in; (void)out_size;

    compute_q_kernel<<<1, 128, 0, stream>>>(query, in_proj_w, in_proj_b, qf);
    cvt_weights_kernel<<<64, 256, 0, stream>>>(in_proj_w, out_w, proj_w, wk, wv, w1, w2);
    init_ws_kernel<<<(GNUM * EDIM) / 256, 256, 0, stream>>>(m_enc, denom, numbuf);

    const int nblk = (N + BN - 1) / BN;
    pass1_kernel<<<nblk, 256, 0, stream>>>(x, seg, qf, in_proj_b, wk, m_enc, logits, N);
    pass2_kernel<<<nblk, 256, 0, stream>>>(x, seg, in_proj_b, wv, m_enc, logits,
                                           denom, numbuf, N);
    pass3_kernel<<<GNUM / BN, 256, 0, stream>>>(numbuf, denom, w1, w2, out_b, proj_b,
                                                (float*)d_out);
}
